// GRU_52578989638422
// MI455X (gfx1250) — matrix operand order repaired and checked by reading, not compiled, not run on hardware
//
#include <hip/hip_runtime.h>
#include <hip/hip_bf16.h>
#include <cstdint>
#include <cstddef>

// ---------------------------------------------------------------------------
// GRU for MI455X (gfx1250, wave32, WMMA + TDM).
//   Phase 1: gates[g] = x @ W_i[g].T + b[g]   (bf16 WMMA 16x16x32, f32 acc,
//            NT stores so the 403MB gate stream doesn't thrash L2)
//   Phase 2: persistent scan kernel, 256 waves (one per 16x16 h-tile),
//            h rows staged into LDS by the Tensor Data Mover (one
//            TENSOR_LOAD_TO_LDS per block per step), gate loads prefetched
//            behind the GEMM, grid barrier per step. Wh (6.3 MB bf16) stays
//            resident in the 192MB L2 for all 512 steps.
// ---------------------------------------------------------------------------

#define GRU_B 64
#define GRU_T 512
#define GRU_I 512
#define GRU_H 1024
#define GRU_M (GRU_B * GRU_T)   // 32768 rows for the input GEMM

typedef __bf16    bf16x16 __attribute__((ext_vector_type(16)));
typedef float     f32x8   __attribute__((ext_vector_type(8)));
typedef unsigned  u32x4   __attribute__((ext_vector_type(4)));
typedef int       i32x4   __attribute__((ext_vector_type(4)));
typedef int       i32x8   __attribute__((ext_vector_type(8)));

union Frag32B { bf16x16 v; u32x4 q[2]; };   // 32 bytes = one WMMA 16-bit A/B fragment

// ---- WMMA wrapper: D = A(16x32 bf16) * B(32x16 bf16) + C(16x16 f32) -------
__device__ __forceinline__ f32x8 wmma_bf16(bf16x16 a, bf16x16 b, f32x8 c) {
    // 8 args: (neg_a, A, neg_b, B, c_mod, C, reuse_a, reuse_b)
    return __builtin_amdgcn_wmma_f32_16x16x32_bf16(false, a, false, b,
                                                   (short)0, c, false, false);
}

// ---- A fragment (row-major [.., ldk] bf16) --------------------------------
// Lane layout (matches B): lanes 0-15 row M=lane, VGPR0..7
// hold K=0..15; lanes 16-31 hold K=16..31.
// => two contiguous 16B loads per lane. Works for global or LDS pointers
// (address space inferred after inlining).
__device__ __forceinline__ bf16x16 load_a_frag(const __bf16* base,
                                               int ldk, int m0, int k0) {
    const int lane = threadIdx.x & 31;
    const int m = m0 + (lane & 15);
    const int kk = (lane < 16) ? 0 : 16;
    const __bf16* p = base + (size_t)m * ldk + (k0 + kk);
    Frag32B f;
    f.q[0] = *reinterpret_cast<const u32x4*>(p);        // K = kk .. kk+7
    f.q[1] = *reinterpret_cast<const u32x4*>(p + 8);    // K = kk+8 .. kk+15
    return f.v;
}

// ---- B fragment from W stored row-major [N, ldk]: B[k][n] = W[n][k] -------
// Lane layout (contiguous k, same as A): lanes 0-15 col N=lane, VGPR0..7 hold K=0..15;
// lanes 16-31 hold K=16..31.  => two contiguous 16B loads per lane.
__device__ __forceinline__ bf16x16 load_b_frag(const __bf16* __restrict__ Wrow,
                                               int ldk, int n0, int k0) {
    const int lane = threadIdx.x & 31;
    const int n = n0 + (lane & 15);
    const int kk = (lane < 16) ? 0 : 16;
    const __bf16* p = Wrow + (size_t)n * ldk + (k0 + kk);
    Frag32B f;
    f.q[0] = *reinterpret_cast<const u32x4*>(p);        // K = kk .. kk+7
    f.q[1] = *reinterpret_cast<const u32x4*>(p + 8);    // K = kk+8 .. kk+15
    return f.v;
}

// Fast gate nonlinearities on the TRANS pipe (v_exp_f32).
__device__ __forceinline__ float fast_sigmoid(float x) {
    return 1.f / (1.f + __expf(-x));
}
__device__ __forceinline__ float fast_tanh(float x) {
    return 2.f / (1.f + __expf(-2.f * x)) - 1.f;
}

// ---------------------------------------------------------------------------
// fp32 -> bf16 conversion (grid-stride)
// ---------------------------------------------------------------------------
__global__ void cvt_f32_bf16(const float* __restrict__ s, __bf16* __restrict__ d,
                             size_t n) {
    size_t i = (size_t)blockIdx.x * blockDim.x + threadIdx.x;
    const size_t stride = (size_t)gridDim.x * blockDim.x;
    for (; i < n; i += stride) d[i] = (__bf16)s[i];
}

// ---------------------------------------------------------------------------
// Phase 1: gates[g][m][n] = sum_k x[m][k] * W_i[g][n][k] + b[g][n]
// One wave computes a 64x16 strip (4 M-tiles sharing each B fragment).
// ---------------------------------------------------------------------------
#define MTILES 4
__global__ void __launch_bounds__(256)
input_gemm(const __bf16* __restrict__ xbf,          // [M, I]
           const __bf16* __restrict__ Wi,           // [3][H][I]
           const float* __restrict__ b0, const float* __restrict__ b1,
           const float* __restrict__ b2,
           float* __restrict__ gates) {             // [3][M][H]
    const int warp = threadIdx.x >> 5;
    const int lane = threadIdx.x & 31;
    const unsigned NT = GRU_H / 16;                     // 64 n-tiles
    const unsigned MS = GRU_M / (16 * MTILES);          // 512 m-strips
    const unsigned wid = blockIdx.x * (blockDim.x >> 5) + warp;
    const unsigned nt = wid % NT;
    const unsigned ms = (wid / NT) % MS;
    const unsigned g  = wid / (NT * MS);
    if (g >= 3) return;

    const int n0 = nt * 16;
    const int m0 = ms * (16 * MTILES);
    const __bf16* Wg = Wi + (size_t)g * GRU_H * GRU_I;
    const float* bp = (g == 0) ? b0 : (g == 1 ? b1 : b2);
    const float bias = bp[n0 + (lane & 15)];

    f32x8 acc[MTILES];
#pragma unroll
    for (int i = 0; i < MTILES; ++i)
#pragma unroll
        for (int r = 0; r < 8; ++r) acc[i][r] = bias;

    for (int k0 = 0; k0 < GRU_I; k0 += 32) {
        const bf16x16 bfrag = load_b_frag(Wg, GRU_I, n0, k0);
#pragma unroll
        for (int i = 0; i < MTILES; ++i) {
            const bf16x16 afrag = load_a_frag(xbf, GRU_I, m0 + 16 * i, k0);
            acc[i] = wmma_bf16(afrag, bfrag, acc[i]);
        }
    }

    // NT stores: 403MB stream written once, read once by the scan -> keep L2
    // free for weights.
    float* gout = gates + (size_t)g * GRU_M * GRU_H;
    const int n = n0 + (lane & 15);
    const int moff = (lane < 16) ? 0 : 8;
#pragma unroll
    for (int i = 0; i < MTILES; ++i)
#pragma unroll
        for (int r = 0; r < 8; ++r)
            __builtin_nontemporal_store(acc[i][r],
                gout + (size_t)(m0 + 16 * i + moff + r) * GRU_H + n);
}

// ---------------------------------------------------------------------------
// Grid-wide barrier (persistent kernel). Counter zeroed per launch via
// hipMemsetAsync; targets grow monotonically within the launch.
// ---------------------------------------------------------------------------
__device__ __forceinline__ void grid_barrier(unsigned* cnt, unsigned nblk,
                                             unsigned step) {
    __threadfence();                 // make h writes device-visible
    __syncthreads();
    if (threadIdx.x == 0) {
        const unsigned target = nblk * (step + 1u);
        __hip_atomic_fetch_add(cnt, 1u, __ATOMIC_RELEASE, __HIP_MEMORY_SCOPE_AGENT);
        while (__hip_atomic_load(cnt, __ATOMIC_ACQUIRE, __HIP_MEMORY_SCOPE_AGENT)
               < target) {
            __builtin_amdgcn_s_sleep(1);
        }
    }
    __syncthreads();
}

// ---------------------------------------------------------------------------
// Phase 2: persistent GRU scan. 32 blocks x 8 waves = 256 waves; wave (mt,nt)
// owns the 16x16 tile (rows 16*mt.., cols 16*nt..) of h. All 8 waves of a
// block share mt, so wave 0 stages the block's 16 h-rows (16x1024 bf16 =
// 32KB, contiguous in hbf) into LDS with ONE Tensor-Data-Mover descriptor
// per step; A fragments then come from LDS (ds_load_b128). Per step: three
// recurrent GEMM accumulators via WMMA over K=1024, then the fused gate
// update. Gate operands are prefetched behind the GEMM.
// ---------------------------------------------------------------------------
__global__ void __launch_bounds__(256, 1)
gru_scan(const float* __restrict__ gates,      // [3][M][H] (bias included)
         const __bf16* __restrict__ Wh,        // [3][H][H]
         float* __restrict__ hf,               // [B][H] fp32 state
         __bf16* __restrict__ hbf,             // [B][H] bf16 mirror (WMMA A)
         float* __restrict__ ys,               // [B][T][H]
         float* __restrict__ hlast,            // [B][H]
         unsigned* __restrict__ sync) {
    __shared__ __bf16 hs[16 * GRU_H];          // 32 KB, LDS offset 0

    const int warp = threadIdx.x >> 5;
    const int lane = threadIdx.x & 31;
    const int wid = blockIdx.x * 8 + warp;     // 0..255
    const int mt = wid >> 6;                   // 0..3   (B/16); == blockIdx>>3
    const int nt = wid & 63;                   // 0..63  (H/16)
    const int m0 = mt * 16, n0 = nt * 16;
    const int n = n0 + (lane & 15);
    const int moff = (lane < 16) ? 0 : 8;

    const float* gR = gates;
    const float* gI = gates + (size_t)GRU_M * GRU_H;
    const float* gN = gates + 2 * (size_t)GRU_M * GRU_H;
    const __bf16* Wr = Wh;
    const __bf16* Wz = Wh + (size_t)GRU_H * GRU_H;
    const __bf16* Wn = Wh + 2 * (size_t)GRU_H * GRU_H;

    // ---- TDM descriptor (ISA 8.3/8.4): 1D tile, 16384 x bf16 = 32KB ------
    // All fields are wave-uniform (derived from blockIdx) -> SGPR-sourced D#.
    const unsigned long long ga =
        (unsigned long long)(const void*)(hbf + (size_t)m0 * GRU_H);
    // group0: count=1 | lds_addr=0 (hs at LDS offset 0) | global_addr | type=2
    const u32x4 g0 = { 1u,
                       0u,
                       (unsigned)(ga & 0xffffffffu),
                       (unsigned)((ga >> 32) & 0x1ffffffu) | (2u << 30) };
    // group1: wg_mask=0 (not in cluster), data_size=1 (2B), tensor_dim0=16384,
    // tensor_dim1=1, tile_dim0=16384 (1D), tile_dim1/2=0, dim0_stride=16384.
    const i32x8 g1 = { (int)(1u << 16),            // data_size = 2 bytes
                       (int)(16384u << 16),        // tensor_dim0[15:0]
                       (int)(1u << 16),            // tensor_dim0 hi=0, dim1 lo=1
                       (int)(16384u << 16),        // dim1 hi=0, tile_dim0=16384
                       0,                          // tile_dim1=0, tile_dim2=0
                       16384,                      // tensor_dim0_stride lo
                       0, 0 };
    const i32x4 gz4 = { 0, 0, 0, 0 };              // groups 2/3: <=2D tensor
    const i32x8 gz8 = { 0, 0, 0, 0, 0, 0, 0, 0 };  // trailing group (clang-23 form)

    for (int t = 0; t < GRU_T; ++t) {
        // ---- Stage h rows m0..m0+15 into LDS via the Tensor Data Mover ---
        if (warp == 0) {
            __builtin_amdgcn_tensor_load_to_lds(g0, g1, gz4, gz4, gz8, 0);
            __builtin_amdgcn_s_wait_tensorcnt(0);
        }
        __syncthreads();
        // The TDM descriptor writes hs behind the compiler's back; escape the
        // pointer so the optimizer must assume hs was modified (keeps the
        // ds_load_b128 fragment reads alive and un-hoisted). Emits nothing.
        asm volatile("" : : "r"(&hs[0]) : "memory");

        // ---- Prefetch this step's gate operands behind the GEMM ----------
#pragma unroll
        for (int r = 0; r < 8; ++r) {
            const size_t gi =
                ((size_t)(m0 + moff + r) * GRU_T + t) * GRU_H + n;
            __builtin_prefetch(gR + gi, 0, 1);
            __builtin_prefetch(gI + gi, 0, 1);
            __builtin_prefetch(gN + gi, 0, 1);
        }

        // ---- Recurrent GEMMs: aG = h @ Wh_G.T for the 16x16 tile ---------
        f32x8 aR = {0.f, 0.f, 0.f, 0.f, 0.f, 0.f, 0.f, 0.f};
        f32x8 aI = aR, aN = aR;
        for (int k0 = 0; k0 < GRU_H; k0 += 32) {
            const bf16x16 a = load_a_frag(hs, GRU_H, 0, k0);   // from LDS
            aR = wmma_bf16(a, load_b_frag(Wr, GRU_H, n0, k0), aR);
            aI = wmma_bf16(a, load_b_frag(Wz, GRU_H, n0, k0), aI);
            aN = wmma_bf16(a, load_b_frag(Wn, GRU_H, n0, k0), aN);
        }

        // ---- Fused gate update ------------------------------------------
#pragma unroll
        for (int r = 0; r < 8; ++r) {
            const int m = m0 + moff + r;                       // batch index
            const size_t gi = ((size_t)m * GRU_T + t) * GRU_H + n;
            const size_t hi = (size_t)m * GRU_H + n;
            const float xr = __builtin_nontemporal_load(gR + gi);
            const float xi = __builtin_nontemporal_load(gI + gi);
            const float xn = __builtin_nontemporal_load(gN + gi);
            const float hv = hf[hi];
            const float rg = fast_sigmoid(xr + aR[r]);
            const float ig = fast_sigmoid(xi + aI[r]);
            const float ng = fast_tanh(xn + rg * aN[r]);
            const float hn = (1.f - ig) * ng + ig * hv;
            hf[hi]  = hn;
            hbf[hi] = (__bf16)hn;
            __builtin_nontemporal_store(hn, ys + gi);
            if (t == GRU_T - 1) hlast[hi] = hn;
        }
        grid_barrier(sync, gridDim.x, (unsigned)t);
    }
}

// ---------------------------------------------------------------------------
// Host-side orchestration (graph-capture safe: only async ops on `stream`).
// ---------------------------------------------------------------------------
extern "C" void kernel_launch(void* const* d_in, const int* in_sizes, int n_in,
                              void* d_out, int out_size, void* d_ws, size_t ws_size,
                              hipStream_t stream) {
    (void)in_sizes; (void)n_in; (void)out_size; (void)ws_size;

    const float* x    = (const float*)d_in[0];
    const float* W_ir = (const float*)d_in[1];
    const float* W_ii = (const float*)d_in[2];
    const float* W_in = (const float*)d_in[3];
    const float* b_ir = (const float*)d_in[4];
    const float* b_ii = (const float*)d_in[5];
    const float* b_in = (const float*)d_in[6];
    const float* W_hr = (const float*)d_in[7];
    const float* W_hi = (const float*)d_in[8];
    const float* W_hn = (const float*)d_in[9];

    float* ys    = (float*)d_out;                               // [B][T][H]
    float* hlast = ys + (size_t)GRU_B * GRU_T * GRU_H;          // [B][H]

    // Workspace layout (all offsets multiples of 256 bytes)
    char* w = (char*)d_ws;
    const size_t SZ_XBF   = (size_t)GRU_M * GRU_I * 2;          // 33.5 MB
    const size_t SZ_WI    = (size_t)3 * GRU_H * GRU_I * 2;      //  3.1 MB
    const size_t SZ_WH    = (size_t)3 * GRU_H * GRU_H * 2;      //  6.3 MB
    const size_t SZ_HF    = (size_t)GRU_B * GRU_H * 4;          //  256 KB
    const size_t SZ_HBF   = (size_t)GRU_B * GRU_H * 2;          //  128 KB
    const size_t SZ_SYNC  = 256;
    __bf16*   xbf   = (__bf16*)(w);
    __bf16*   Wi_bf = (__bf16*)(w + SZ_XBF);
    __bf16*   Wh_bf = (__bf16*)(w + SZ_XBF + SZ_WI);
    float*    hf    = (float*) (w + SZ_XBF + SZ_WI + SZ_WH);
    __bf16*   hbf   = (__bf16*)(w + SZ_XBF + SZ_WI + SZ_WH + SZ_HF);
    unsigned* sync  = (unsigned*)(w + SZ_XBF + SZ_WI + SZ_WH + SZ_HF + SZ_HBF);
    float*    gates = (float*) (w + SZ_XBF + SZ_WI + SZ_WH + SZ_HF + SZ_HBF + SZ_SYNC);

    // Zero the state + barrier counter (poisoned 0xAA by harness).
    (void)hipMemsetAsync(hf,   0, SZ_HF,   stream);
    (void)hipMemsetAsync(hbf,  0, SZ_HBF,  stream);
    (void)hipMemsetAsync(sync, 0, SZ_SYNC, stream);

    // fp32 -> bf16 conversions
    const size_t HI = (size_t)GRU_H * GRU_I;   // 524288
    const size_t HH = (size_t)GRU_H * GRU_H;   // 1048576
    cvt_f32_bf16<<<2048, 256, 0, stream>>>(x,    xbf,           (size_t)GRU_M * GRU_I);
    cvt_f32_bf16<<<512,  256, 0, stream>>>(W_ir, Wi_bf,          HI);
    cvt_f32_bf16<<<512,  256, 0, stream>>>(W_ii, Wi_bf + HI,     HI);
    cvt_f32_bf16<<<512,  256, 0, stream>>>(W_in, Wi_bf + 2 * HI, HI);
    cvt_f32_bf16<<<1024, 256, 0, stream>>>(W_hr, Wh_bf,          HH);
    cvt_f32_bf16<<<1024, 256, 0, stream>>>(W_hi, Wh_bf + HH,     HH);
    cvt_f32_bf16<<<1024, 256, 0, stream>>>(W_hn, Wh_bf + 2 * HH, HH);

    // Phase 1: 3 gates x 512 m-strips x 64 n-tiles = 98304 wave-tasks,
    // 8 waves per 256-thread block -> 12288 blocks.
    input_gemm<<<12288, 256, 0, stream>>>(xbf, Wi_bf, b_ir, b_ii, b_in, gates);

    // Phase 2: persistent scan. 32 blocks x 256 threads = 256 waves (one per
    // 16x16 tile of the [64,1024] hidden state).
    gru_scan<<<32, 256, 0, stream>>>(gates, Wh_bf, hf, hbf, ys, hlast, sync);
}
